// Attention_90460601189287
// MI455X (gfx1250) — matrix operand order repaired and checked by reading, not compiled, not run on hardware
//
#include <hip/hip_runtime.h>
#include <math.h>

// ---------------------------------------------------------------- types
typedef __attribute__((ext_vector_type(16))) __bf16 v16bf;
typedef __attribute__((ext_vector_type(8)))  __bf16 v8bf;
typedef __attribute__((ext_vector_type(8)))  float  v8f;

#define DIMX   1024
#define HEADS  16
#define HD     64
#define BATCH  2
#define SEQ    2048
#define ROWS   (BATCH*SEQ)   // 4096
#define QKV_N  (3*DIMX)      // 3072

__device__ __forceinline__ unsigned short f2bf(float f) {
  unsigned int x = __float_as_uint(f);
  unsigned int r = x + 0x7FFFu + ((x >> 16) & 1u);   // round-to-nearest-even
  return (unsigned short)(r >> 16);
}

union AF { v16bf v; v8bf h[2]; };

// build a 16-element bf16 fragment from two contiguous 8-element (16B) chunks
__device__ __forceinline__ v16bf load_frag_2x8(const unsigned short* p0,
                                               const unsigned short* p1) {
  AF f;
  f.h[0] = *reinterpret_cast<const v8bf*>(p0);
  f.h[1] = *reinterpret_cast<const v8bf*>(p1);
  return f.v;
}

// ---------------------------------------------------------------- small helpers (sigma)
__global__ void matvec_rows(const float* __restrict__ W, const float* __restrict__ x,
                            float* __restrict__ y, int R, int C) {
  __shared__ float red[256];
  int r = blockIdx.x;
  float acc = 0.f;
  for (int c = threadIdx.x; c < C; c += blockDim.x) acc += W[(size_t)r * C + c] * x[c];
  red[threadIdx.x] = acc; __syncthreads();
  for (int s = 128; s > 0; s >>= 1) {
    if ((int)threadIdx.x < s) red[threadIdx.x] += red[threadIdx.x + s];
    __syncthreads();
  }
  if (threadIdx.x == 0) y[r] = red[0];
}

__global__ void matvec_cols(const float* __restrict__ W, const float* __restrict__ x,
                            float* __restrict__ y, int R, int C) {
  __shared__ float red[256];
  int c = blockIdx.x;
  float acc = 0.f;
  for (int r = threadIdx.x; r < R; r += blockDim.x) acc += W[(size_t)r * C + c] * x[r];
  red[threadIdx.x] = acc; __syncthreads();
  for (int s = 128; s > 0; s >>= 1) {
    if ((int)threadIdx.x < s) red[threadIdx.x] += red[threadIdx.x + s];
    __syncthreads();
  }
  if (threadIdx.x == 0) y[c] = red[0];
}

__global__ void normalize_vec(const float* __restrict__ x, float* __restrict__ y, int n) {
  __shared__ float red[256];
  float acc = 0.f;
  for (int i = threadIdx.x; i < n; i += blockDim.x) acc += x[i] * x[i];
  red[threadIdx.x] = acc; __syncthreads();
  for (int s = 128; s > 0; s >>= 1) {
    if ((int)threadIdx.x < s) red[threadIdx.x] += red[threadIdx.x + s];
    __syncthreads();
  }
  float inv = rsqrtf(red[0]);
  for (int i = threadIdx.x; i < n; i += blockDim.x) y[i] = x[i] * inv;
}

__global__ void dot_scale(const float* __restrict__ a, const float* __restrict__ b,
                          int n, const float* __restrict__ sigma_param,
                          float* __restrict__ out_scale) {
  __shared__ float red[256];
  float acc = 0.f;
  for (int i = threadIdx.x; i < n; i += blockDim.x) acc += a[i] * b[i];
  red[threadIdx.x] = acc; __syncthreads();
  for (int s = 128; s > 0; s >>= 1) {
    if ((int)threadIdx.x < s) red[threadIdx.x] += red[threadIdx.x + s];
    __syncthreads();
  }
  if (threadIdx.x == 0) out_scale[0] = sigma_param[0] / red[0];
}

// ---------------------------------------------------------------- conversions
__global__ void cvt_bf16(const float* __restrict__ src, unsigned short* __restrict__ dst, int n) {
  int i = blockIdx.x * blockDim.x + threadIdx.x;
  if (i < n) dst[i] = f2bf(src[i]);
}

// W [K,N] fp32 row-major  ->  Wt [N,K] bf16 row-major
__global__ void transpose_bf16(const float* __restrict__ W, unsigned short* __restrict__ Wt,
                               int K, int N) {
  int i = blockIdx.x * blockDim.x + threadIdx.x;
  if (i >= K * N) return;
  int n = i / K, k = i - n * K;
  Wt[(size_t)n * K + k] = f2bf(W[(size_t)k * N + n]);
}

// Vt[((b*H+h)*64 + d)*SEQ + n] = qkv_bf16[(b*SEQ+n)*3072 + 2048 + h*64 + d]
__global__ void build_vt(const unsigned short* __restrict__ qkv,
                         unsigned short* __restrict__ Vt, int total) {
  int i = blockIdx.x * blockDim.x + threadIdx.x;
  if (i >= total) return;
  int n  = i & (SEQ - 1);
  int d  = (i >> 11) & 63;
  int bh = i >> 17;
  int b = bh >> 4, h = bh & 15;
  Vt[i] = qkv[(size_t)(b * SEQ + n) * QKV_N + 2 * DIMX + h * HD + d];
}

// ---------------------------------------------------------------- WMMA GEMM (register-blocked)
// C[M,N] = scale * A[M,K] @ B ;  B provided transposed: Bt[N,K].  bf16 in, fp32 acc.
// 8 waves / block in a 2(M) x 4(N) arrangement. Each wave owns a 32x64 tile:
//   2 A-fragments x 4 B-fragments -> 8 WMMAs per 32-wide k-step.
// Block tile = 64(M) x 256(N).  Loads per WMMA: 1.5 x b128 (vs 8 in naive form).
__global__ void __launch_bounds__(256)
gemm_bf16_wmma(const unsigned short* __restrict__ A,
               const unsigned short* __restrict__ Bt,
               void* __restrict__ Cout,
               const float* __restrict__ scale_p,
               int M, int N, int K, int out_f32)
{
  const int lane = threadIdx.x & 31;
  const int wave = threadIdx.x >> 5;
  const int half = lane >> 4;
  const int lid  = lane & 15;
  const int wrow = wave >> 2;          // 0..1
  const int wcol = wave & 3;           // 0..3

  const int m0 = blockIdx.y * 64 + wrow * 32;
  const int n0 = blockIdx.x * 256 + wcol * 64;

  const unsigned short* arow0 = A  + (size_t)(m0 + lid) * K;
  const unsigned short* arow1 = A  + (size_t)(m0 + 16 + lid) * K;
  const unsigned short* brow0 = Bt + (size_t)(n0 + lid) * K;
  const unsigned short* brow1 = Bt + (size_t)(n0 + 16 + lid) * K;
  const unsigned short* brow2 = Bt + (size_t)(n0 + 32 + lid) * K;
  const unsigned short* brow3 = Bt + (size_t)(n0 + 48 + lid) * K;

  v8f acc[2][4] = {};
  for (int k0 = 0; k0 < K; k0 += 32) {
    // prefetch the streamed A rows 8 k-steps (256 elements) ahead (global_prefetch_b8 path)
    __builtin_prefetch(arow0 + k0 + 256, 0, 0);
    __builtin_prefetch(arow1 + k0 + 256, 0, 0);

    v16bf af0 = load_frag_2x8(arow0 + k0 + half * 16, arow0 + k0 + half * 16 + 8);
    v16bf af1 = load_frag_2x8(arow1 + k0 + half * 16, arow1 + k0 + half * 16 + 8);
    v16bf bf0 = load_frag_2x8(brow0 + k0 + half * 16, brow0 + k0 + half * 16 + 8);
    v16bf bf1 = load_frag_2x8(brow1 + k0 + half * 16, brow1 + k0 + half * 16 + 8);
    v16bf bf2 = load_frag_2x8(brow2 + k0 + half * 16, brow2 + k0 + half * 16 + 8);
    v16bf bf3 = load_frag_2x8(brow3 + k0 + half * 16, brow3 + k0 + half * 16 + 8);

    acc[0][0] = __builtin_amdgcn_wmma_f32_16x16x32_bf16(false, af0, false, bf0, (short)0, acc[0][0], false, false);
    acc[0][1] = __builtin_amdgcn_wmma_f32_16x16x32_bf16(false, af0, false, bf1, (short)0, acc[0][1], false, false);
    acc[0][2] = __builtin_amdgcn_wmma_f32_16x16x32_bf16(false, af0, false, bf2, (short)0, acc[0][2], false, false);
    acc[0][3] = __builtin_amdgcn_wmma_f32_16x16x32_bf16(false, af0, false, bf3, (short)0, acc[0][3], false, false);
    acc[1][0] = __builtin_amdgcn_wmma_f32_16x16x32_bf16(false, af1, false, bf0, (short)0, acc[1][0], false, false);
    acc[1][1] = __builtin_amdgcn_wmma_f32_16x16x32_bf16(false, af1, false, bf1, (short)0, acc[1][1], false, false);
    acc[1][2] = __builtin_amdgcn_wmma_f32_16x16x32_bf16(false, af1, false, bf2, (short)0, acc[1][2], false, false);
    acc[1][3] = __builtin_amdgcn_wmma_f32_16x16x32_bf16(false, af1, false, bf3, (short)0, acc[1][3], false, false);
  }

  const float s = scale_p[0];
#pragma unroll
  for (int i = 0; i < 2; ++i) {
#pragma unroll
    for (int t = 0; t < 4; ++t) {
      const int col = n0 + t * 16 + lid;
      if (out_f32) {
        float* C = (float*)Cout;
#pragma unroll
        for (int r = 0; r < 8; ++r)
          C[(size_t)(m0 + i * 16 + r + half * 8) * N + col] = acc[i][t][r] * s;
      } else {
        unsigned short* C = (unsigned short*)Cout;
#pragma unroll
        for (int r = 0; r < 8; ++r)
          C[(size_t)(m0 + i * 16 + r + half * 8) * N + col] = f2bf(acc[i][t][r] * s);
      }
    }
  }
}

// ---------------------------------------------------------------- fused causal attention
// grid.x = SEQ/64 query blocks, grid.y = B*H.  4 waves/block, wave = 16 query rows.
__global__ void __launch_bounds__(128)
flash_attn_wmma(const unsigned short* __restrict__ qkv,   // [ROWS, 3072] bf16
                const unsigned short* __restrict__ Vt,    // [B*H, 64, SEQ] bf16
                unsigned short* __restrict__ Oout)        // [ROWS, 1024] bf16
{
  __shared__ unsigned short pbuf[4][16][32];   // per-wave P tile (C-layout -> A-layout bounce)

  const int lane = threadIdx.x & 31;
  const int wave = threadIdx.x >> 5;
  const int half = lane >> 4;
  const int lid  = lane & 15;

  const int qb = blockIdx.x * 64;
  const int bh = blockIdx.y;
  const int b  = bh >> 4;
  const int h  = bh & 15;
  const int qbase = qb + wave * 16;

  // Q fragments for this wave's 16 rows (K = 64 split into two 32-chunks)
  const unsigned short* qrow = qkv + (size_t)(b * SEQ + qbase + lid) * QKV_N + h * HD;
  const v16bf qa0 = load_frag_2x8(qrow + half * 16,      qrow + half * 16 + 8);
  const v16bf qa1 = load_frag_2x8(qrow + 32 + half * 16, qrow + 32 + half * 16 + 8);

  v8f o[4] = {};                    // 16q x 64d accumulator (4 C-frags)
  float mrow[8], lrow[8];
#pragma unroll
  for (int r = 0; r < 8; ++r) { mrow[r] = -3.0e38f; lrow[r] = 0.0f; }

  const unsigned short* kbase = qkv + (size_t)(b * SEQ) * QKV_N + DIMX + h * HD;
  const unsigned short* vbase = Vt + (size_t)bh * HD * SEQ;

  for (int kt = 0; kt <= qbase + 15; kt += 32) {          // wave-uniform causal bound
    // ---- S = Q K^T for two 16-key tiles
    v8f s[2];
#pragma unroll
    for (int t = 0; t < 2; ++t) {
      const unsigned short* krow = kbase + (size_t)(kt + t * 16 + lid) * QKV_N;
      v16bf kb0 = load_frag_2x8(krow + half * 16,      krow + half * 16 + 8);
      v16bf kb1 = load_frag_2x8(krow + 32 + half * 16, krow + 32 + half * 16 + 8);
      v8f acc = {};
      acc = __builtin_amdgcn_wmma_f32_16x16x32_bf16(false, qa0, false, kb0,
                                                    (short)0, acc, false, false);
      acc = __builtin_amdgcn_wmma_f32_16x16x32_bf16(false, qa1, false, kb1,
                                                    (short)0, acc, false, false);
      s[t] = acc;
    }
    // ---- scale + causal mask (data mask, EXEC stays all-ones)
#pragma unroll
    for (int t = 0; t < 2; ++t) {
      const int kcol = kt + t * 16 + lid;
#pragma unroll
      for (int r = 0; r < 8; ++r) {
        const int qrow_i = qbase + r + half * 8;
        float v = s[t][r] * 0.125f;                 // hd^-0.5
        s[t][r] = (kcol > qrow_i) ? -3.0e38f : v;
      }
    }
    // ---- online softmax (row stats live per half-wave per C-layout)
    float fs[8];
#pragma unroll
    for (int r = 0; r < 8; ++r) {
      float rm = fmaxf(s[0][r], s[1][r]);
#pragma unroll
      for (int m = 1; m < 16; m <<= 1) rm = fmaxf(rm, __shfl_xor(rm, m, 32));
      float nm = fmaxf(mrow[r], rm);
      float f  = __expf(mrow[r] - nm);
      float p0 = __expf(s[0][r] - nm);
      float p1 = __expf(s[1][r] - nm);
      s[0][r] = p0; s[1][r] = p1;
      float rs = p0 + p1;
#pragma unroll
      for (int m = 1; m < 16; m <<= 1) rs += __shfl_xor(rs, m, 32);
      lrow[r] = lrow[r] * f + rs;
      mrow[r] = nm;
      fs[r] = f;
    }
#pragma unroll
    for (int r = 0; r < 8; ++r) {
      o[0][r] *= fs[r]; o[1][r] *= fs[r]; o[2][r] *= fs[r]; o[3][r] *= fs[r];
    }
    // ---- P: C-layout -> A-fragment layout via per-wave LDS bounce
#pragma unroll
    for (int r = 0; r < 8; ++r) {
      pbuf[wave][r + half * 8][lid]      = f2bf(s[0][r]);
      pbuf[wave][r + half * 8][16 + lid] = f2bf(s[1][r]);
    }
    asm volatile("s_wait_dscnt 0" ::: "memory");   // DS in-order per wave; stores visible
    const unsigned short* prow = &pbuf[wave][lid][0];
    v16bf pa = load_frag_2x8(prow + half * 16, prow + half * 16 + 8);
    // ---- O += P V   (V read via pre-transposed Vt: contiguous fragments)
#pragma unroll
    for (int dt = 0; dt < 4; ++dt) {
      const unsigned short* vrow = vbase + (size_t)(dt * 16 + lid) * SEQ + kt + half * 16;
      v16bf vb = load_frag_2x8(vrow, vrow + 8);
      o[dt] = __builtin_amdgcn_wmma_f32_16x16x32_bf16(false, pa, false, vb,
                                                      (short)0, o[dt], false, false);
    }
  }

  // ---- finalize: divide by row sums, write bf16 [ROWS, DIM] (head-concat layout)
#pragma unroll
  for (int r = 0; r < 8; ++r) {
    const float inv = 1.0f / lrow[r];
    unsigned short* orow = Oout + (size_t)(b * SEQ + qbase + r + half * 8) * DIMX + h * HD;
    orow[0 * 16 + lid] = f2bf(o[0][r] * inv);
    orow[1 * 16 + lid] = f2bf(o[1][r] * inv);
    orow[2 * 16 + lid] = f2bf(o[2][r] * inv);
    orow[3 * 16 + lid] = f2bf(o[3][r] * inv);
  }
}

// ---------------------------------------------------------------- driver
extern "C" void kernel_launch(void* const* d_in, const int* in_sizes, int n_in,
                              void* d_out, int out_size, void* d_ws, size_t ws_size,
                              hipStream_t stream) {
  const float* batch      = (const float*)d_in[0];   // [2,2048,1024]
  const float* W_qkv      = (const float*)d_in[1];   // [1024,3072]
  const float* u_qkv      = (const float*)d_in[2];   // [3072]
  const float* sigma_qkv  = (const float*)d_in[3];   // [1]
  const float* W_proj     = (const float*)d_in[4];   // [1024,1024]
  const float* u_proj     = (const float*)d_in[5];   // [1024]
  const float* sigma_proj = (const float*)d_in[6];   // [1]
  float* out = (float*)d_out;

  char* ws = (char*)d_ws;
  size_t off = 0;
  auto alloc = [&](size_t bytes) { char* p = ws + off; off += (bytes + 255) & ~(size_t)255; return p; };

  unsigned short* Xb     = (unsigned short*)alloc((size_t)ROWS * DIMX * 2);   // 8 MB
  unsigned short* WqkvT  = (unsigned short*)alloc((size_t)QKV_N * DIMX * 2);  // 6 MB
  unsigned short* WprojT = (unsigned short*)alloc((size_t)DIMX * DIMX * 2);   // 2 MB
  unsigned short* qkvb   = (unsigned short*)alloc((size_t)ROWS * QKV_N * 2);  // 24 MB
  unsigned short* Vt     = (unsigned short*)alloc((size_t)ROWS * DIMX * 2);   // 8 MB
  unsigned short* attnb  = (unsigned short*)alloc((size_t)ROWS * DIMX * 2);   // 8 MB
  float* vecA   = (float*)alloc(QKV_N * 4);
  float* vecB   = (float*)alloc(QKV_N * 4);
  float* vecC   = (float*)alloc(QKV_N * 4);
  float* scaleQ = (float*)alloc(4);
  float* scaleP = (float*)alloc(4);
  (void)ws_size; (void)n_in; (void)in_sizes; (void)out_size;

  // ---- sigma for W_qkv (one power-iteration step, as in reference)
  matvec_rows<<<1024, 256, 0, stream>>>(W_qkv, u_qkv, vecA, 1024, QKV_N);
  normalize_vec<<<1, 256, 0, stream>>>(vecA, vecB, 1024);              // v
  matvec_cols<<<QKV_N, 256, 0, stream>>>(W_qkv, vecB, vecA, 1024, QKV_N);
  normalize_vec<<<1, 256, 0, stream>>>(vecA, vecC, QKV_N);             // u2
  matvec_rows<<<1024, 256, 0, stream>>>(W_qkv, vecC, vecA, 1024, QKV_N);
  dot_scale<<<1, 256, 0, stream>>>(vecB, vecA, 1024, sigma_qkv, scaleQ);

  // ---- sigma for W_proj
  matvec_rows<<<1024, 256, 0, stream>>>(W_proj, u_proj, vecA, 1024, DIMX);
  normalize_vec<<<1, 256, 0, stream>>>(vecA, vecB, 1024);
  matvec_cols<<<1024, 256, 0, stream>>>(W_proj, vecB, vecA, 1024, DIMX);
  normalize_vec<<<1, 256, 0, stream>>>(vecA, vecC, DIMX);
  matvec_rows<<<1024, 256, 0, stream>>>(W_proj, vecC, vecA, 1024, DIMX);
  dot_scale<<<1, 256, 0, stream>>>(vecB, vecA, 1024, sigma_proj, scaleP);

  // ---- bf16 conversions / transposes
  cvt_bf16<<<(ROWS * DIMX + 255) / 256, 256, 0, stream>>>(batch, Xb, ROWS * DIMX);
  transpose_bf16<<<(DIMX * QKV_N + 255) / 256, 256, 0, stream>>>(W_qkv, WqkvT, DIMX, QKV_N);
  transpose_bf16<<<(DIMX * DIMX + 255) / 256, 256, 0, stream>>>(W_proj, WprojT, DIMX, DIMX);

  // ---- QKV projection: qkvb = scaleQ * X @ W_qkv   (bf16 out)
  gemm_bf16_wmma<<<dim3(QKV_N / 256, ROWS / 64), 256, 0, stream>>>(
      Xb, WqkvT, qkvb, scaleQ, ROWS, QKV_N, DIMX, 0);

  // ---- V transpose for PV GEMM fragments
  build_vt<<<(ROWS * DIMX + 255) / 256, 256, 0, stream>>>(qkvb, Vt, ROWS * DIMX);

  // ---- fused causal attention
  flash_attn_wmma<<<dim3(SEQ / 64, BATCH * HEADS), 128, 0, stream>>>(qkvb, Vt, attnb);

  // ---- output projection: out = scaleP * attn @ W_proj  (fp32 out)
  gemm_bf16_wmma<<<dim3(DIMX / 256, ROWS / 64), 256, 0, stream>>>(
      attnb, WprojT, out, scaleP, ROWS, DIMX, DIMX, 1);
}
